// SparseMatrixLayer_52518860095721
// MI455X (gfx1250) — matrix operand order repaired and checked by reading, not compiled, not run on hardware
//
#include <hip/hip_runtime.h>
#include <hip/hip_bf16.h>

// ---------------------------------------------------------------------------
// SparseMatrixLayer on MI455X (gfx1250):
//   1) densify COO -> W fp32 (scatter atomicAdd, duplicates summed)
//   2) split W (and X, if workspace allows) into bf16 hi/lo planes
//   3) GEMM out[m,n] = sum_k X[m,k]*W[n,k] via v_wmma_f32_16x16x32_bf16,
//      fp32 emulated as hi*hi + hi*lo + lo*hi (3 bf16 WMMAs per K=32 step)
//   Fast path: double-buffered LDS staged with GLOBAL_LOAD_ASYNC_TO_LDS_B128
// ---------------------------------------------------------------------------

#define COUT 1024
#define CIN  1024
#define NNZ  131072
#define MROWS 16384          // 4 * 4096

#define BM 128
#define BN 128
#define BK 32
#define BKP 40               // padded LDS row (elements) -> 80B stride, breaks bank conflicts

typedef __attribute__((ext_vector_type(16))) __bf16 v16bf;
typedef __attribute__((ext_vector_type(8)))  __bf16 bf16x8;
typedef __attribute__((ext_vector_type(4)))  __bf16 bf16x4;
typedef __attribute__((ext_vector_type(8)))  float  v8f;
typedef __attribute__((ext_vector_type(4)))  float  f32x4;

// Builtin-matching 16-byte vector type (per hipcc diagnostic: the async-LDS
// builtin takes 'int __attribute__((__vector_size__(16)))' pointers).
typedef int i32x4v __attribute__((__vector_size__(16)));

// ---------------- CDNA5 async global->LDS path (probe via __has_builtin) ----
#if defined(__has_builtin)
#if __has_builtin(__builtin_amdgcn_global_load_async_to_lds_b128)
#define ASYNC_LDS 1
#endif
#endif
#ifndef ASYNC_LDS
#define ASYNC_LDS 0
#endif

static __device__ inline void copy16B_to_lds(__bf16* lds_dst, const __bf16* gsrc) {
#if ASYNC_LDS
    __builtin_amdgcn_global_load_async_to_lds_b128(
        (__attribute__((address_space(1))) i32x4v*)gsrc,
        (__attribute__((address_space(3))) i32x4v*)lds_dst,
        /*imm offset=*/0, /*cpol=*/0);
#else
    *(bf16x8*)lds_dst = *(const bf16x8*)gsrc;
#endif
}

static __device__ inline void wait_async_lds() {
#if ASYNC_LDS
#if __has_builtin(__builtin_amdgcn_s_wait_asynccnt)
    __builtin_amdgcn_s_wait_asynccnt(0);
#else
    asm volatile("s_wait_asynccnt 0x0" ::: "memory");
#endif
#endif
}

// ---------------------------------------------------------------------------
static __device__ inline v8f wmma_bf16(v16bf a, v16bf b, v8f c) {
    // (neg_a, A, neg_b, B, c_mod, C, reuse_a, reuse_b)
    return __builtin_amdgcn_wmma_f32_16x16x32_bf16(false, a, false, b,
                                                   (short)0, c, false, false);
}

static __device__ inline v16bf cat16(bf16x8 a, bf16x8 b) {
    return __builtin_shufflevector(a, b, 0, 1, 2, 3, 4, 5, 6, 7,
                                         8, 9, 10, 11, 12, 13, 14, 15);
}

// Shared inner compute: A/B fragment gathers + 24 WMMAs for one K=32 step.
static __device__ inline void compute_step(const __bf16* sXhi, const __bf16* sXlo,
                                           const __bf16* sWhi, const __bf16* sWlo,
                                           int waveM, int waveN, int l15, int lh,
                                           v8f acc[2][4]) {
    // A fragments: 16-bit 16x32 layout.
    // lane l15 = row; VGPR0-3 hold k = lh*8 + 0..7, VGPR4-7 hold k = 16 + lh*8 + 0..7
    v16bf aHi[2], aLo[2];
#pragma unroll
    for (int im = 0; im < 2; ++im) {
        int r = waveM * 32 + im * 16 + l15;
        const __bf16* ph = sXhi + r * BKP + lh * 8;
        const __bf16* pl = sXlo + r * BKP + lh * 8;
        aHi[im] = cat16(*(const bf16x8*)(ph), *(const bf16x8*)(ph + 16));
        aLo[im] = cat16(*(const bf16x8*)(pl), *(const bf16x8*)(pl + 16));
    }
    // B fragments: 32x16 layout; lane l15 = column n; k = lh*8 + 0..7, then 16 + lh*8 + 0..7
#pragma unroll
    for (int in = 0; in < 4; ++in) {
        int c = waveN * 64 + in * 16 + l15;
        const __bf16* ph = sWhi + c * BKP + lh * 8;
        const __bf16* pl = sWlo + c * BKP + lh * 8;
        v16bf bHi = cat16(*(const bf16x8*)(ph), *(const bf16x8*)(ph + 16));
        v16bf bLo = cat16(*(const bf16x8*)(pl), *(const bf16x8*)(pl + 16));
#pragma unroll
        for (int im = 0; im < 2; ++im) {
            acc[im][in] = wmma_bf16(aHi[im], bHi, acc[im][in]);  // hi*hi
            acc[im][in] = wmma_bf16(aHi[im], bLo, acc[im][in]);  // hi*lo
            acc[im][in] = wmma_bf16(aLo[im], bHi, acc[im][in]);  // lo*hi
        }
    }
}

// Epilogue: C/D layout: VGPR r -> row (r + 8*lh), col l15
static __device__ inline void store_acc(float* __restrict__ Out, int m0, int n0,
                                        int waveM, int waveN, int l15, int lh,
                                        v8f acc[2][4]) {
#pragma unroll
    for (int im = 0; im < 2; ++im) {
        int mrow = m0 + waveM * 32 + im * 16 + lh * 8;
#pragma unroll
        for (int in = 0; in < 4; ++in) {
            int ncol = n0 + waveN * 64 + in * 16 + l15;
#pragma unroll
            for (int r = 0; r < 8; ++r) {
                Out[(size_t)(mrow + r) * COUT + ncol] = acc[im][in][r];
            }
        }
    }
}

// ---------------------------------------------------------------------------
// Phase 1: zero the fp32 W accumulator in workspace
// ---------------------------------------------------------------------------
__global__ void zero_w_kernel(float* __restrict__ W) {
    W[(size_t)blockIdx.x * 256 + threadIdx.x] = 0.0f;
}

// ---------------------------------------------------------------------------
// Phase 2: COO scatter-add (coalesce semantics: duplicates sum)
// ---------------------------------------------------------------------------
__global__ void scatter_kernel(const float* __restrict__ vals,
                               const int* __restrict__ rows,
                               const int* __restrict__ cols,
                               float* __restrict__ W) {
    int i = blockIdx.x * 256 + threadIdx.x;
    if (i < NNZ) {
        atomicAdd(W + (size_t)rows[i] * CIN + cols[i], vals[i]);
    }
}

// ---------------------------------------------------------------------------
// Phase 3: split fp32 -> bf16 hi/lo planes (hi = rne(f), lo = rne(f - hi))
// ---------------------------------------------------------------------------
__global__ void split_kernel(const float* __restrict__ S,
                             __bf16* __restrict__ Hi,
                             __bf16* __restrict__ Lo) {
    size_t i = (size_t)blockIdx.x * 256 + threadIdx.x;
    float f  = S[i];
    __bf16 h = (__bf16)f;
    Hi[i]    = h;
    Lo[i]    = (__bf16)(f - (float)h);
}

// ---------------------------------------------------------------------------
// Phase 4a (fast path): all-bf16-plane GEMM, double-buffered async LDS staging
// ---------------------------------------------------------------------------
__global__ __launch_bounds__(256)
void gemm_planes_kernel(const __bf16* __restrict__ Xhi, const __bf16* __restrict__ Xlo,
                        const __bf16* __restrict__ Whi, const __bf16* __restrict__ Wlo,
                        float* __restrict__ Out) {
    __shared__ __bf16 sXhi[2][BM * BKP];
    __shared__ __bf16 sXlo[2][BM * BKP];
    __shared__ __bf16 sWhi[2][BN * BKP];
    __shared__ __bf16 sWlo[2][BN * BKP];

    const int K = CIN;
    const int m0   = blockIdx.y * BM;
    const int n0   = blockIdx.x * BN;
    const int tid  = threadIdx.x;
    const int lane = tid & 31;
    const int wid  = tid >> 5;
    const int waveM = wid & 3;   // 4 waves along M: 32 rows each
    const int waveN = wid >> 2;  // 2 waves along N: 64 cols each
    const int l15  = lane & 15;
    const int lh   = lane >> 4;

    v8f acc[2][4];
#pragma unroll
    for (int im = 0; im < 2; ++im)
#pragma unroll
        for (int in = 0; in < 4; ++in)
            acc[im][in] = (v8f){0.f, 0.f, 0.f, 0.f, 0.f, 0.f, 0.f, 0.f};

    // Stage one K=32 tile set (4 planes) into LDS buffer `buf`:
    // per plane: 128 rows x 32 bf16 = 512 x 16B chunks -> 2 chunks/thread.
    auto stage = [&](int buf, int k0) {
#pragma unroll
        for (int it = 0; it < 2; ++it) {
            int idx = tid + it * 256;          // 0..511
            int row = idx >> 2;                // 128 rows
            int c8  = (idx & 3) * 8;           // 4 chunks of 8 bf16 (16B)
            int l   = row * BKP + c8;
            size_t gx = (size_t)(m0 + row) * K + k0 + c8;
            size_t gw = (size_t)(n0 + row) * K + k0 + c8;
            copy16B_to_lds(&sXhi[buf][l], Xhi + gx);
            copy16B_to_lds(&sXlo[buf][l], Xlo + gx);
            copy16B_to_lds(&sWhi[buf][l], Whi + gw);
            copy16B_to_lds(&sWlo[buf][l], Wlo + gw);
        }
    };

    stage(0, 0);
    int buf = 0;
    for (int k0 = 0; k0 < K; k0 += BK) {
        wait_async_lds();       // this wave's async tile writes have landed
        __syncthreads();        // everyone's landed; prior buffer reads retired
        if (k0 + BK < K) stage(buf ^ 1, k0 + BK);   // overlap next staging with compute
        compute_step(sXhi[buf], sXlo[buf], sWhi[buf], sWlo[buf],
                     waveM, waveN, l15, lh, acc);
        buf ^= 1;
    }
    store_acc(Out, m0, n0, waveM, waveN, l15, lh, acc);
}

// ---------------------------------------------------------------------------
// Phase 4b (fallback, small workspace): convert X fp32->bf16 pair in-kernel
// ---------------------------------------------------------------------------
__global__ __launch_bounds__(256)
void gemm_bf16x3_kernel(const float* __restrict__ X,
                        const __bf16* __restrict__ Whi,
                        const __bf16* __restrict__ Wlo,
                        float* __restrict__ Out) {
    __shared__ __bf16 sXhi[BM * BKP];
    __shared__ __bf16 sXlo[BM * BKP];
    __shared__ __bf16 sWhi[BN * BKP];
    __shared__ __bf16 sWlo[BN * BKP];

    const int K = CIN;
    const int m0   = blockIdx.y * BM;
    const int n0   = blockIdx.x * BN;
    const int tid  = threadIdx.x;
    const int lane = tid & 31;
    const int wid  = tid >> 5;
    const int waveM = wid & 3;
    const int waveN = wid >> 2;
    const int l15  = lane & 15;
    const int lh   = lane >> 4;

    v8f acc[2][4];
#pragma unroll
    for (int im = 0; im < 2; ++im)
#pragma unroll
        for (int in = 0; in < 4; ++in)
            acc[im][in] = (v8f){0.f, 0.f, 0.f, 0.f, 0.f, 0.f, 0.f, 0.f};

    for (int k0 = 0; k0 < K; k0 += BK) {
        // stage X tile (fp32 -> bf16 hi/lo), 1024 float4 chunks
#pragma unroll
        for (int it = 0; it < 4; ++it) {
            int idx = tid + it * 256;
            int row = idx >> 3;
            int c4  = (idx & 7) * 4;
            f32x4 xv = *(const f32x4*)(X + (size_t)(m0 + row) * K + k0 + c4);
            bf16x4 hv, lv;
#pragma unroll
            for (int e = 0; e < 4; ++e) {
                float  f = xv[e];
                __bf16 h = (__bf16)f;
                hv[e] = h;
                lv[e] = (__bf16)(f - (float)h);
            }
            *(bf16x4*)(sXhi + row * BKP + c4) = hv;
            *(bf16x4*)(sXlo + row * BKP + c4) = lv;
        }
        // stage W bf16 planes (async path when available)
#pragma unroll
        for (int it = 0; it < 2; ++it) {
            int idx = tid + it * 256;
            int row = idx >> 2;
            int c8  = (idx & 3) * 8;
            copy16B_to_lds(sWhi + row * BKP + c8, Whi + (size_t)(n0 + row) * K + k0 + c8);
            copy16B_to_lds(sWlo + row * BKP + c8, Wlo + (size_t)(n0 + row) * K + k0 + c8);
        }
        wait_async_lds();
        __syncthreads();
        compute_step(sXhi, sXlo, sWhi, sWlo, waveM, waveN, l15, lh, acc);
        __syncthreads();
    }
    store_acc(Out, m0, n0, waveM, waveN, l15, lh, acc);
}

// ---------------------------------------------------------------------------
// Host-side launcher
// ---------------------------------------------------------------------------
extern "C" void kernel_launch(void* const* d_in, const int* in_sizes, int n_in,
                              void* d_out, int out_size, void* d_ws, size_t ws_size,
                              hipStream_t stream) {
    const float* x      = (const float*)d_in[0];   // [4,4096,1024] fp32
    const float* vals   = (const float*)d_in[1];   // [NNZ]
    const int*   idx    = (const int*)d_in[2];     // [2, NNZ]: rows then cols
    const int*   rows   = idx;
    const int*   cols   = idx + NNZ;
    float*       out    = (float*)d_out;           // [16384, 1024] fp32

    // Workspace layout:
    //   W fp32 (4 MB) | Whi (2 MB) | Wlo (2 MB) | [ Xhi (32 MB) | Xlo (32 MB) ]
    const size_t nW = (size_t)COUT * CIN;          // 1M
    const size_t nX = (size_t)MROWS * CIN;         // 16M
    char*   ws  = (char*)d_ws;
    float*  W   = (float*)(ws);
    __bf16* Whi = (__bf16*)(ws + nW * sizeof(float));
    __bf16* Wlo = Whi + nW;
    __bf16* Xhi = (__bf16*)(ws + nW * (sizeof(float) + 2 * sizeof(__bf16)));
    __bf16* Xlo = Xhi + nX;
    const size_t need_planes = nW * (sizeof(float) + 2 * sizeof(__bf16))
                             + nX * 2 * sizeof(__bf16);          // 72 MB

    // Phase 1: zero W (deterministic every call; d_ws is not re-poisoned)
    zero_w_kernel<<<(int)(nW / 256), 256, 0, stream>>>(W);
    // Phase 2: COO scatter-add densify
    scatter_kernel<<<NNZ / 256, 256, 0, stream>>>(vals, rows, cols, W);
    // Phase 3: split W into bf16 hi/lo planes
    split_kernel<<<(int)(nW / 256), 256, 0, stream>>>(W, Whi, Wlo);

    dim3 grid(COUT / BN, MROWS / BM);
    if (ws_size >= need_planes) {
        // Pre-split X once (64 MB extra HBM traffic ~5.5us @23.3TB/s), then a
        // pure-bf16, zero-conversion, async-double-buffered WMMA GEMM.
        split_kernel<<<(int)(nX / 256), 256, 0, stream>>>(x, Xhi, Xlo);
        gemm_planes_kernel<<<grid, 256, 0, stream>>>(Xhi, Xlo, Whi, Wlo, out);
    } else {
        gemm_bf16x3_kernel<<<grid, 256, 0, stream>>>(x, Whi, Wlo, out);
    }
}
